// DeltaNet_31877247271555
// MI455X (gfx1250) — compile-verified, matrix operand order checked by reading, not run on hardware
//
#include <hip/hip_runtime.h>
#include <hip/hip_bf16.h>

typedef _Float16 half_t;
typedef __attribute__((ext_vector_type(16))) _Float16 v16h;
typedef __attribute__((ext_vector_type(8)))  _Float16 v8h;
typedef __attribute__((ext_vector_type(8)))  float    v8f;
typedef __attribute__((ext_vector_type(4)))  int      v4i;

#define DEV __device__ __forceinline__

static constexpr int Bb   = 4;
static constexpr int Ll   = 4096;
static constexpr int Dd   = 1024;
static constexpr int Hh   = 4;
static constexpr int Dk   = 256;
static constexpr int Dv   = 256;
static constexpr int Mrows = Bb * Ll;          // 16384
static constexpr int Kgate = 1056;             // 1052 padded to mult of 32
static constexpr int Ngate = 2048;

DEV float sigmoidf_(float x) { return 1.f / (1.f + __expf(-x)); }
DEV float siluf_(float x)    { return x * sigmoidf_(x); }
DEV float geluf_(float x)    { return 0.5f * x * (1.f + erff(x * 0.70710678118f)); }

// ---------------------------------------------------------------------------
// CDNA5 async global->LDS copy (GLOBAL_LOAD_ASYNC_TO_LDS_B128, ASYNCcnt).
// Builtin signature (from hipcc diagnostic): (int4 as1*, int4 as3*, Ii, Ii).
// ---------------------------------------------------------------------------
#if defined(__has_builtin)
#if __has_builtin(__builtin_amdgcn_global_load_async_to_lds_b128) && \
    __has_builtin(__builtin_amdgcn_s_wait_asynccnt)
#define USE_ASYNC_LDS 1
#endif
#endif

#ifdef USE_ASYNC_LDS
DEV void async_cp16(void* lds, const void* g) {
  // LDS generic addresses carry the LDS byte offset in the low 32 bits
  // (aperture check, ISA 10.2), so integer truncation yields a valid
  // addrspace(3) pointer; global generic addresses are the raw VA.
  __builtin_amdgcn_global_load_async_to_lds_b128(
      (__attribute__((address_space(1))) v4i*)(uintptr_t)g,
      (__attribute__((address_space(3))) v4i*)(uint32_t)(uintptr_t)lds,
      0, 0);
}
DEV void async_wait0() { __builtin_amdgcn_s_wait_asynccnt(0); }
#endif

// ---------------------------------------------------------------------------
// WMMA fragment helpers (gfx1250 wave32, 16x16x32 f16 -> f32)
// A (16x32): lanes 0-15 row=lane K in {0..7,16..23}; lanes 16-31 row=lane-16,
// K in {8..15,24..31}.  B (32x16): same with lane->column.  C: vgpr r ->
// row r (+8 for upper lanes), lane&15 -> column.
// ---------------------------------------------------------------------------
DEV v8f wmma_f16(v16h a, v16h b, v8f c) {
  return __builtin_amdgcn_wmma_f32_16x16x32_f16(false, a, false, b,
                                                (short)0, c, false, false);
}

// A fragment from row-major f16 (row stride ld elements); contiguous K runs
DEV v16h frag_a(const half_t* base, int ld) {
  int lane = threadIdx.x & 31;
  int row  = lane & 15;
  int kh   = (lane >> 4) * 8;
  v8h lo = *(const v8h*)(base + row * ld + kh);
  v8h hi = *(const v8h*)(base + row * ld + 16 + kh);
  v16h f;
#pragma unroll
  for (int e = 0; e < 8; e++) { f[e] = lo[e]; f[8 + e] = hi[e]; }
  return f;
}

// B fragment from K-major f16 (B[k][n], row stride ld); lane = column
DEV v16h frag_b(const half_t* base, int ld) {
  int lane = threadIdx.x & 31;
  int col  = lane & 15;
  int kh   = (lane >> 4) * 8;
  v16h f;
#pragma unroll
  for (int e = 0; e < 8; e++) {
    f[e]     = base[(kh + e) * ld + col];
    f[8 + e] = base[(16 + kh + e) * ld + col];
  }
  return f;
}

// B fragment from K-major f32 source (converted to f16 on load)
DEV v16h frag_b_f32(const float* base, int ld) {
  int lane = threadIdx.x & 31;
  int col  = lane & 15;
  int kh   = (lane >> 4) * 8;
  v16h f;
#pragma unroll
  for (int e = 0; e < 8; e++) {
    f[e]     = (half_t)base[(kh + e) * ld + col];
    f[8 + e] = (half_t)base[(16 + kh + e) * ld + col];
  }
  return f;
}

// A fragment from f32 source (attn matrix)
DEV v16h frag_a_f32(const float* base, int ld) {
  int lane = threadIdx.x & 31;
  int row  = lane & 15;
  int kh   = (lane >> 4) * 8;
  v16h f;
#pragma unroll
  for (int e = 0; e < 8; e++) {
    f[e]     = (half_t)base[row * ld + kh + e];
    f[8 + e] = (half_t)base[row * ld + 16 + kh + e];
  }
  return f;
}

// A fragment where the logical A is the transpose of the stored f16 matrix:
// A[m][k] = base[k*ld + m]
DEV v16h frag_aT(const half_t* base, int ld) {
  int lane = threadIdx.x & 31;
  int row  = lane & 15;
  int kh   = (lane >> 4) * 8;
  v16h f;
#pragma unroll
  for (int e = 0; e < 8; e++) {
    f[e]     = base[(kh + e) * ld + row];
    f[8 + e] = base[(16 + kh + e) * ld + row];
  }
  return f;
}

// ---------------------------------------------------------------------------
// Generic WMMA GEMM: C[M,N] = A[M,K] (f16, row major) * B[K,N] (f16, K major)
// EPI: 0 = f32 out, 1 = f16 out, 2 = (+bias, GELU) f16 out
// BM=128 BN=64 BK=32, 256 threads = 8 waves, wave tile 32x32 (2x2 subtiles)
// LDS staging uses GLOBAL_LOAD_ASYNC_TO_LDS_B128 when available.
// ---------------------------------------------------------------------------
template <int EPI>
__global__ __launch_bounds__(256) void gemm_wmma(
    const half_t* __restrict__ A, const half_t* __restrict__ B,
    const float* __restrict__ bias, void* __restrict__ Cout,
    int M, int N, int K) {
  constexpr int BM = 128, BN = 64, BK = 32;
  __shared__ half_t As[BM * BK];
  __shared__ half_t Bs[BK * BN];
  const int tid = threadIdx.x, wave = tid >> 5, lane = tid & 31;
  const int m0 = blockIdx.y * BM, n0 = blockIdx.x * BN;
  const int wm = (wave & 3) * 32, wn = (wave >> 2) * 32;
  v8f acc[2][2] = {};

  for (int k0 = 0; k0 < K; k0 += BK) {
#ifdef USE_ASYNC_LDS
#pragma unroll
    for (int i = 0; i < 2; i++) {  // 512 x b128 for A tile
      int u = tid * 2 + i, r = u >> 2, c = (u & 3) * 8;
      async_cp16(As + r * BK + c, A + (size_t)(m0 + r) * K + k0 + c);
    }
    {  // 256 x b128 for B tile
      int r = tid >> 3, c = (tid & 7) * 8;
      async_cp16(Bs + r * BN + c, B + (size_t)(k0 + r) * N + n0 + c);
    }
    async_wait0();
#else
#pragma unroll
    for (int i = 0; i < 2; i++) {  // 512 uint4 for A tile
      int u = tid * 2 + i, r = u >> 2, c = (u & 3) * 8;
      *(uint4*)(As + r * BK + c) =
          *(const uint4*)(A + (size_t)(m0 + r) * K + k0 + c);
    }
    {  // 256 uint4 for B tile
      int r = tid >> 3, c = (tid & 7) * 8;
      *(uint4*)(Bs + r * BN + c) =
          *(const uint4*)(B + (size_t)(k0 + r) * N + n0 + c);
    }
#endif
    __syncthreads();
#pragma unroll
    for (int mi = 0; mi < 2; mi++) {
      v16h a = frag_a(As + (wm + mi * 16) * BK, BK);
#pragma unroll
      for (int ni = 0; ni < 2; ni++) {
        v16h b = frag_b(Bs + wn + ni * 16, BN);
        acc[mi][ni] = wmma_f16(a, b, acc[mi][ni]);
      }
    }
    __syncthreads();
  }

  const int col = lane & 15, ro = (lane >> 4) * 8;
#pragma unroll
  for (int mi = 0; mi < 2; mi++)
#pragma unroll
    for (int ni = 0; ni < 2; ni++)
#pragma unroll
      for (int r = 0; r < 8; r++) {
        int gm = m0 + wm + mi * 16 + r + ro;
        int gn = n0 + wn + ni * 16 + col;
        float v = acc[mi][ni][r];
        if (bias) v += bias[gn];
        if constexpr (EPI == 0) {
          ((float*)Cout)[(size_t)gm * N + gn] = v;
        } else if constexpr (EPI == 1) {
          ((half_t*)Cout)[(size_t)gm * N + gn] = (half_t)v;
        } else {
          ((half_t*)Cout)[(size_t)gm * N + gn] = (half_t)geluf_(v);
        }
      }
}

// ---------------------------------------------------------------------------
// Elementwise conversion kernels
// ---------------------------------------------------------------------------
__global__ void f32_to_f16_kernel(const float* __restrict__ in,
                                  half_t* __restrict__ out, size_t n) {
  size_t i = (size_t)blockIdx.x * 256 + threadIdx.x;
  if (i < n) out[i] = (half_t)in[i];
}

__global__ void gw1_pad_kernel(const float* __restrict__ w,
                               half_t* __restrict__ out) {
  size_t i = (size_t)blockIdx.x * 256 + threadIdx.x;
  if (i >= (size_t)Kgate * Ngate) return;
  int r = (int)(i / Ngate), c = (int)(i % Ngate);
  out[i] = (r < 1052) ? (half_t)w[(size_t)r * Ngate + c] : (half_t)0.f;
}

// ---------------------------------------------------------------------------
// beta = sigmoid(hidden @ w_b)   (one block per row)
// ---------------------------------------------------------------------------
__global__ __launch_bounds__(256) void beta_kernel(
    const float* __restrict__ hid, const float* __restrict__ wb,
    float* __restrict__ beta) {
  __shared__ float acc[4];
  int m = blockIdx.x, t = threadIdx.x;
  if (t < 4) acc[t] = 0.f;
  __syncthreads();
  float loc[4] = {0.f, 0.f, 0.f, 0.f};
  for (int d = t; d < Dd; d += 256) {
    float x = hid[(size_t)m * Dd + d];
#pragma unroll
    for (int j = 0; j < 4; j++) loc[j] += x * wb[d * 4 + j];
  }
#pragma unroll
  for (int j = 0; j < 4; j++) atomicAdd(&acc[j], loc[j]);
  __syncthreads();
  if (t < 4) beta[(size_t)m * 4 + t] = sigmoidf_(acc[t]);
}

// ---------------------------------------------------------------------------
// Causal depthwise conv (k=4) + SiLU.  F32OUT selects f32 output (for v).
// ---------------------------------------------------------------------------
template <bool F32OUT>
__global__ void dwconv4_silu(const half_t* __restrict__ x,
                             const float* __restrict__ w,
                             void* __restrict__ out) {
  size_t idx = (size_t)blockIdx.x * 256 + threadIdx.x;
  if (idx >= (size_t)Mrows * Dd) return;
  int d = (int)(idx & (Dd - 1));
  size_t m = idx >> 10;
  int l = (int)(m & (Ll - 1));
  float acc = 0.f;
#pragma unroll
  for (int t = 0; t < 4; t++) {
    int ll = l - 3 + t;
    if (ll >= 0) acc += w[d * 4 + t] * (float)x[(m - (size_t)(3 - t)) * Dd + d];
  }
  float y = siluf_(acc);
  if constexpr (F32OUT) ((float*)out)[idx] = y;
  else                  ((half_t*)out)[idx] = (half_t)y;
}

// ---------------------------------------------------------------------------
// In-place per-head L2 norm over Dk (one block per (b,l) row)
// ---------------------------------------------------------------------------
__global__ __launch_bounds__(256) void l2norm_heads(half_t* __restrict__ x) {
  __shared__ float ssq[4];
  int m = blockIdx.x, t = threadIdx.x;
  if (t < 4) ssq[t] = 0.f;
  __syncthreads();
  float v[4];
#pragma unroll
  for (int h = 0; h < 4; h++) {
    v[h] = (float)x[(size_t)m * Dd + h * Dk + t];
    atomicAdd(&ssq[h], v[h] * v[h]);
  }
  __syncthreads();
#pragma unroll
  for (int h = 0; h < 4; h++) {
    float s = rsqrtf(ssq[h] + 1e-6f);
    x[(size_t)m * Dd + h * Dk + t] = (half_t)(v[h] * s);
  }
}

// ---------------------------------------------------------------------------
// Chunkwise delta rule.  One block = one (b,h, dv-slice of 64).  The state
// slice S (256x64 f32) lives in LDS for the whole 128-chunk scan.
// ---------------------------------------------------------------------------
static constexpr size_t DELTA_SMEM =
    256 * 64 * 4 +          // S
    3 * 32 * 256 * 2 +      // qt kt wt
    3 * 32 * 64 * 2 +       // ut vt ua
    32 * 32 * 4 +           // At
    32 * 4;                 // beta

__global__ __launch_bounds__(256) void delta_kernel(
    const half_t* __restrict__ qn, const half_t* __restrict__ kn,
    const float* __restrict__ vcf, const float* __restrict__ beta,
    float* __restrict__ dout) {
  extern __shared__ char smem[];
  float*  S   = (float*)smem;                         // 256 x 64
  half_t* qt  = (half_t*)(smem + 256 * 64 * 4);       // 32 x 256
  half_t* kt  = qt + 32 * 256;
  half_t* wt  = kt + 32 * 256;
  half_t* ut  = wt + 32 * 256;                        // 32 x 64
  half_t* vt  = ut + 32 * 64;
  half_t* ua  = vt + 32 * 64;
  float*  At  = (float*)(ua + 32 * 64);               // 32 x 32
  float*  bet = At + 32 * 32;

  const int tid = threadIdx.x, wave = tid >> 5, lane = tid & 31;
  const int vs = blockIdx.x & 3, bh = blockIdx.x >> 2;
  const int h = bh & 3, b = bh >> 2;

  const half_t* qb = qn  + (size_t)b * Ll * Dd + h * Dk;
  const half_t* kb = kn  + (size_t)b * Ll * Dd + h * Dk;
  const float*  vb = vcf + (size_t)b * Ll * Dd + h * Dv + vs * 64;
  float*        ob = dout + (size_t)b * Ll * Dd + h * Dv + vs * 64;
  const float*  bb = beta + (size_t)b * Ll * 4 + h;

  for (int i = tid; i < 256 * 64; i += 256) S[i] = 0.f;
  __syncthreads();

  const int col = lane & 15, ro = (lane >> 4) * 8;

  for (int c = 0; c < 128; c++) {
    const int l0 = c * 32;
    // ---- stage q,k,beta,v*beta tiles ----
#ifdef USE_ASYNC_LDS
    for (int u = tid; u < 1024; u += 256) {
      int r = u >> 5, cc = (u & 31) * 8;
      async_cp16(qt + r * 256 + cc, qb + (size_t)(l0 + r) * Dd + cc);
      async_cp16(kt + r * 256 + cc, kb + (size_t)(l0 + r) * Dd + cc);
    }
#else
    for (int u = tid; u < 1024; u += 256) {
      int r = u >> 5, cc = (u & 31) * 8;
      *(uint4*)(qt + r * 256 + cc) =
          *(const uint4*)(qb + (size_t)(l0 + r) * Dd + cc);
      *(uint4*)(kt + r * 256 + cc) =
          *(const uint4*)(kb + (size_t)(l0 + r) * Dd + cc);
    }
#endif
    if (tid < 32) bet[tid] = bb[(size_t)(l0 + tid) * 4];
    // prefetch next chunk's q/k rows (global_prefetch_b8)
    if (c + 1 < 128) {
      int pr = tid >> 3, pc = (tid & 7) * 32;
      __builtin_prefetch(qb + (size_t)(l0 + 32 + pr) * Dd + pc, 0, 0);
      __builtin_prefetch(kb + (size_t)(l0 + 32 + pr) * Dd + pc, 0, 0);
    }
#ifdef USE_ASYNC_LDS
    async_wait0();
#endif
    __syncthreads();
    for (int i = tid; i < 32 * 64; i += 256) {
      int r = i >> 6, cc = i & 63;
      vt[i] = (half_t)(vb[(size_t)(l0 + r) * Dd + cc] * bet[r]);
    }
    // ---- A = (beta .* k) @ k^T  (waves 0..3, one 16x16 tile each) ----
    if (wave < 4) {
      int mt = (wave >> 1) & 1, nt = wave & 1;
      v8f acc = {};
      half_t bs = (half_t)bet[mt * 16 + (lane & 15)];
      for (int kk = 0; kk < 8; kk++) {
        v16h a = frag_a(kt + mt * 16 * 256 + kk * 32, 256);
#pragma unroll
        for (int e = 0; e < 16; e++) a[e] = a[e] * bs;
        // B = k^T : read kt with lane as column of the result
        v16h bbq = frag_a(kt + nt * 16 * 256 + kk * 32, 256);
        acc = wmma_f16(a, bbq, acc);
      }
#pragma unroll
      for (int r = 0; r < 8; r++)
        At[(mt * 16 + r + ro) * 32 + nt * 16 + col] = acc[r];
    }
    __syncthreads();
    // ---- forward substitution: M = I + tril(A,-1); w = M^-1 (beta.*k),
    //      u = M^-1 (beta.*v) ----
    for (int i = 0; i < 32; i++) {
      {
        float x = bet[i] * (float)kt[i * 256 + tid];
        for (int j = 0; j < i; j++)
          x -= At[i * 32 + j] * (float)wt[j * 256 + tid];
        wt[i * 256 + tid] = (half_t)x;
      }
      if (tid < 64) {
        float x = (float)vt[i * 64 + tid];
        for (int j = 0; j < i; j++)
          x -= At[i * 32 + j] * (float)ut[j * 64 + tid];
        ut[i * 64 + tid] = (half_t)x;
      }
      __syncthreads();
    }
    // ---- attn = tril(q @ k^T) into At (reuse) ----
    if (wave < 4) {
      int mt = (wave >> 1) & 1, nt = wave & 1;
      v8f acc = {};
      for (int kk = 0; kk < 8; kk++) {
        v16h a  = frag_a(qt + mt * 16 * 256 + kk * 32, 256);
        v16h bq = frag_a(kt + nt * 16 * 256 + kk * 32, 256);
        acc = wmma_f16(a, bq, acc);
      }
#pragma unroll
      for (int r = 0; r < 8; r++) {
        int gm = mt * 16 + r + ro, gn = nt * 16 + col;
        At[gm * 32 + gn] = (gn <= gm) ? acc[r] : 0.f;
      }
    }
    __syncthreads();
    // ---- u_adj = u - w @ S  (8 waves, 2x4 tiles) ----
    {
      int mt = wave & 1, nt = wave >> 1;
      v8f acc = {};
      for (int kk = 0; kk < 8; kk++) {
        v16h a = frag_a(wt + mt * 16 * 256 + kk * 32, 256);
        v16h bq = frag_b_f32(S + kk * 32 * 64 + nt * 16, 64);
        acc = wmma_f16(a, bq, acc);
      }
#pragma unroll
      for (int r = 0; r < 8; r++) {
        int rr = mt * 16 + r + ro, cc = nt * 16 + col;
        ua[rr * 64 + cc] = (half_t)((float)ut[rr * 64 + cc] - acc[r]);
      }
    }
    __syncthreads();
    // ---- out = q @ S + attn @ u_adj ----
    {
      int mt = wave & 1, nt = wave >> 1;
      v8f acc = {};
      for (int kk = 0; kk < 8; kk++) {
        v16h a = frag_a(qt + mt * 16 * 256 + kk * 32, 256);
        v16h bq = frag_b_f32(S + kk * 32 * 64 + nt * 16, 64);
        acc = wmma_f16(a, bq, acc);
      }
      v16h a2 = frag_a_f32(At + mt * 16 * 32, 32);
      v16h b2 = frag_b(ua + nt * 16, 64);
      acc = wmma_f16(a2, b2, acc);
#pragma unroll
      for (int r = 0; r < 8; r++) {
        int rr = mt * 16 + r + ro, cc = nt * 16 + col;
        ob[(size_t)(l0 + rr) * Dd + cc] = acc[r];
      }
    }
    __syncthreads();
    // ---- S += k^T @ u_adj  (64 tiles: 16 m-tiles x 4 n-tiles) ----
    for (int t = wave * 8; t < wave * 8 + 8; t++) {
      int mt = t >> 2, nt = t & 3;
      v16h a = frag_aT(kt + mt * 16, 256);   // A[m][kk] = kt[kk*256 + m]
      v16h bq = frag_b(ua + nt * 16, 64);
      v8f acc;
#pragma unroll
      for (int r = 0; r < 8; r++)
        acc[r] = S[(mt * 16 + r + ro) * 64 + nt * 16 + col];
      acc = wmma_f16(a, bq, acc);
#pragma unroll
      for (int r = 0; r < 8; r++)
        S[(mt * 16 + r + ro) * 64 + nt * 16 + col] = acc[r];
    }
    __syncthreads();
  }
}

// ---------------------------------------------------------------------------
// FIR taps helper
// ---------------------------------------------------------------------------
template <int KW>
DEV float fir_apply(const float* __restrict__ w, int d, const float* x) {
  float y = 0.f;
#pragma unroll
  for (int t = 0; t < KW; t++) y += w[d * KW + (KW - 1 - t)] * x[t];
  return y;
}

// ---------------------------------------------------------------------------
// Stream stats: mean|fir_k(v)| (k=1,3,7,15,31), mean|delta|, mean|v| per head.
// stats layout: stats[m*28 + s*4 + h]
// ---------------------------------------------------------------------------
__global__ __launch_bounds__(256) void stats_kernel(
    const float* __restrict__ vcf, const float* __restrict__ dout,
    const float* __restrict__ f1, const float* __restrict__ f3,
    const float* __restrict__ f7, const float* __restrict__ f15,
    const float* __restrict__ f31, float* __restrict__ stats) {
  __shared__ float acc[28];
  const int m = blockIdx.x, l = m & (Ll - 1), t = threadIdx.x;
  if (t < 28) acc[t] = 0.f;
  __syncthreads();
  for (int h = 0; h < 4; h++) {
    int d = h * Dv + t;
    float x[31];
#pragma unroll
    for (int tau = 0; tau < 31; tau++)
      x[tau] = (tau <= l) ? vcf[((size_t)m - tau) * Dd + d] : 0.f;
    float s0 = fabsf(fir_apply<1>(f1, d, x));
    float s1 = fabsf(fir_apply<3>(f3, d, x));
    float s2 = fabsf(fir_apply<7>(f7, d, x));
    float s3 = fabsf(fir_apply<15>(f15, d, x));
    float s4 = fabsf(fir_apply<31>(f31, d, x));
    float s5 = fabsf(dout[(size_t)m * Dd + d]);
    float s6 = fabsf(x[0]);
    atomicAdd(&acc[0 * 4 + h], s0);
    atomicAdd(&acc[1 * 4 + h], s1);
    atomicAdd(&acc[2 * 4 + h], s2);
    atomicAdd(&acc[3 * 4 + h], s3);
    atomicAdd(&acc[4 * 4 + h], s4);
    atomicAdd(&acc[5 * 4 + h], s5);
    atomicAdd(&acc[6 * 4 + h], s6);
  }
  __syncthreads();
  if (t < 28) stats[(size_t)m * 28 + t] = acc[t] * (1.f / 256.f);
}

// ---------------------------------------------------------------------------
// gate input: [hidden(1024) | stats(28) | zeros(4)] as f16 (K = 1056)
// ---------------------------------------------------------------------------
__global__ void gi_kernel(const float* __restrict__ hid,
                          const float* __restrict__ stats,
                          half_t* __restrict__ gi) {
  size_t idx = (size_t)blockIdx.x * 256 + threadIdx.x;
  if (idx >= (size_t)Mrows * Kgate) return;
  size_t m = idx / Kgate;
  int j = (int)(idx % Kgate);
  float v = (j < Dd) ? hid[m * Dd + j]
                     : ((j < Dd + 28) ? stats[m * 28 + (j - Dd)] : 0.f);
  gi[idx] = (half_t)v;
}

// ---------------------------------------------------------------------------
// gate layer 2 (N=28) + softplus + per-head normalization -> wts[m*28+h*7+s]
// ---------------------------------------------------------------------------
__global__ __launch_bounds__(256) void gate2_kernel(
    const half_t* __restrict__ hmid, const float* __restrict__ w2,
    const float* __restrict__ b2, float* __restrict__ wts) {
  __shared__ float acc[28];
  __shared__ float wv[28];
  const int m = blockIdx.x, t = threadIdx.x;
  if (t < 28) acc[t] = 0.f;
  __syncthreads();
  float loc[28];
#pragma unroll
  for (int j = 0; j < 28; j++) loc[j] = 0.f;
  for (int k = t; k < Ngate; k += 256) {
    float x = (float)hmid[(size_t)m * Ngate + k];
#pragma unroll
    for (int j = 0; j < 28; j++) loc[j] += x * w2[(size_t)k * 28 + j];
  }
#pragma unroll
  for (int j = 0; j < 28; j++) atomicAdd(&acc[j], loc[j]);
  __syncthreads();
  if (t < 28) {
    float xg = acc[t] + b2[t];
    float sp = (xg > 20.f) ? xg : log1pf(__expf(xg));
    wv[t] = sp + 0.02f;
  }
  __syncthreads();
  if (t < 28) {
    int h = t / 7;
    float s = 0.f;
#pragma unroll
    for (int j = 0; j < 7; j++) s += wv[h * 7 + j];
    wts[(size_t)m * 28 + t] = wv[t] / s;
  }
}

// ---------------------------------------------------------------------------
// Combine streams (recomputing FIRs) + RMS norm -> o_h (f16)
// ---------------------------------------------------------------------------
__global__ __launch_bounds__(256) void combine_kernel(
    const float* __restrict__ vcf, const float* __restrict__ dout,
    const float* __restrict__ wts,
    const float* __restrict__ f1, const float* __restrict__ f3,
    const float* __restrict__ f7, const float* __restrict__ f15,
    const float* __restrict__ f31, const float* __restrict__ norm_w,
    half_t* __restrict__ oh) {
  __shared__ float obuf[1024];
  __shared__ float ssq[4];
  const int m = blockIdx.x, l = m & (Ll - 1), t = threadIdx.x;
  if (t < 4) ssq[t] = 0.f;
  __syncthreads();
  for (int h = 0; h < 4; h++) {
    int d = h * Dv + t;
    float x[31];
#pragma unroll
    for (int tau = 0; tau < 31; tau++)
      x[tau] = (tau <= l) ? vcf[((size_t)m - tau) * Dd + d] : 0.f;
    float y0 = fir_apply<1>(f1, d, x);
    float y1 = fir_apply<3>(f3, d, x);
    float y2 = fir_apply<7>(f7, d, x);
    float y3 = fir_apply<15>(f15, d, x);
    float y4 = fir_apply<31>(f31, d, x);
    float y5 = dout[(size_t)m * Dd + d];
    float y6 = x[0];
    const float* w7 = wts + (size_t)m * 28 + h * 7;
    float o = w7[0] * y0 + w7[1] * y1 + w7[2] * y2 + w7[3] * y3 +
              w7[4] * y4 + w7[5] * y5 + w7[6] * y6;
    obuf[d] = o;
    atomicAdd(&ssq[h], o * o);
  }
  __syncthreads();
  for (int h = 0; h < 4; h++) {
    int d = h * Dv + t;
    float r = rsqrtf(ssq[h] * (1.f / 256.f) + 1e-5f);
    oh[(size_t)m * Dd + d] = (half_t)(obuf[d] * r * norm_w[t]);
  }
}

// ---------------------------------------------------------------------------
// Host-side launcher
// ---------------------------------------------------------------------------
extern "C" void kernel_launch(void* const* d_in, const int* in_sizes, int n_in,
                              void* d_out, int out_size, void* d_ws,
                              size_t ws_size, hipStream_t stream) {
  const float* hid   = (const float*)d_in[0];
  const float* w_q   = (const float*)d_in[1];
  const float* w_k   = (const float*)d_in[2];
  const float* w_v   = (const float*)d_in[3];
  const float* w_b   = (const float*)d_in[4];
  const float* cq    = (const float*)d_in[5];
  const float* ck    = (const float*)d_in[6];
  const float* cv    = (const float*)d_in[7];
  const float* f1    = (const float*)d_in[8];
  const float* f3    = (const float*)d_in[9];
  const float* f7    = (const float*)d_in[10];
  const float* f15   = (const float*)d_in[11];
  const float* f31   = (const float*)d_in[12];
  const float* gw1   = (const float*)d_in[13];
  const float* gb1   = (const float*)d_in[14];
  const float* gw2   = (const float*)d_in[15];
  const float* gb2   = (const float*)d_in[16];
  const float* nw    = (const float*)d_in[17];
  const float* w_o   = (const float*)d_in[18];
  float* out = (float*)d_out;

  // workspace arena
  char* ws = (char*)d_ws;
  size_t off = 0;
  auto alloc = [&](size_t bytes) {
    size_t o = off;
    off = (off + bytes + 255) & ~(size_t)255;
    return o;
  };
  const size_t nMD = (size_t)Mrows * Dd;
  half_t* hid_h = (half_t*)(ws + alloc(nMD * 2));
  half_t* wq_h  = (half_t*)(ws + alloc((size_t)Dd * Dd * 2));
  half_t* wk_h  = (half_t*)(ws + alloc((size_t)Dd * Dd * 2));
  half_t* wv_h  = (half_t*)(ws + alloc((size_t)Dd * Dd * 2));
  half_t* wo_h  = (half_t*)(ws + alloc((size_t)Dd * Dd * 2));
  half_t* gw1_h = (half_t*)(ws + alloc((size_t)Kgate * Ngate * 2));
  half_t* ql_h  = (half_t*)(ws + alloc(nMD * 2));
  half_t* kl_h  = (half_t*)(ws + alloc(nMD * 2));
  half_t* vl_h  = (half_t*)(ws + alloc(nMD * 2));
  half_t* qc_h  = (half_t*)(ws + alloc(nMD * 2));   // conv+silu then l2norm
  half_t* kc_h  = (half_t*)(ws + alloc(nMD * 2));
  float*  vc_f  = (float*)(ws + alloc(nMD * 4));
  float*  betab = (float*)(ws + alloc((size_t)Mrows * 4 * 4));
  float*  dob   = (float*)(ws + alloc(nMD * 4));    // delta out
  float*  stats = (float*)(ws + alloc((size_t)Mrows * 28 * 4));
  half_t* gi_h  = (half_t*)(ws + alloc((size_t)Mrows * Kgate * 2));
  half_t* hm_h  = (half_t*)(ws + alloc((size_t)Mrows * Ngate * 2));
  float*  wts   = (float*)(ws + alloc((size_t)Mrows * 28 * 4));
  half_t* o_h   = (half_t*)(ws + alloc(nMD * 2));
  (void)ws_size; (void)n_in; (void)in_sizes; (void)out_size;

  auto blocks = [](size_t n) { return (unsigned)((n + 255) / 256); };

  // 1) conversions
  f32_to_f16_kernel<<<blocks(nMD), 256, 0, stream>>>(hid, hid_h, nMD);
  f32_to_f16_kernel<<<blocks((size_t)Dd * Dd), 256, 0, stream>>>(w_q, wq_h, (size_t)Dd * Dd);
  f32_to_f16_kernel<<<blocks((size_t)Dd * Dd), 256, 0, stream>>>(w_k, wk_h, (size_t)Dd * Dd);
  f32_to_f16_kernel<<<blocks((size_t)Dd * Dd), 256, 0, stream>>>(w_v, wv_h, (size_t)Dd * Dd);
  f32_to_f16_kernel<<<blocks((size_t)Dd * Dd), 256, 0, stream>>>(w_o, wo_h, (size_t)Dd * Dd);
  gw1_pad_kernel<<<blocks((size_t)Kgate * Ngate), 256, 0, stream>>>(gw1, gw1_h);

  // 2) q/k/v projections (WMMA, f16 out)
  dim3 gqkv(Dd / 64, Mrows / 128);
  gemm_wmma<1><<<gqkv, 256, 0, stream>>>(hid_h, wq_h, nullptr, ql_h, Mrows, Dd, Dd);
  gemm_wmma<1><<<gqkv, 256, 0, stream>>>(hid_h, wk_h, nullptr, kl_h, Mrows, Dd, Dd);
  gemm_wmma<1><<<gqkv, 256, 0, stream>>>(hid_h, wv_h, nullptr, vl_h, Mrows, Dd, Dd);

  // 3) beta
  beta_kernel<<<Mrows, 256, 0, stream>>>(hid, w_b, betab);

  // 4) conv + silu; l2norm on q,k
  dwconv4_silu<false><<<blocks(nMD), 256, 0, stream>>>(ql_h, cq, qc_h);
  dwconv4_silu<false><<<blocks(nMD), 256, 0, stream>>>(kl_h, ck, kc_h);
  dwconv4_silu<true ><<<blocks(nMD), 256, 0, stream>>>(vl_h, cv, vc_f);
  l2norm_heads<<<Mrows, 256, 0, stream>>>(qc_h);
  l2norm_heads<<<Mrows, 256, 0, stream>>>(kc_h);

  // 5) delta rule scan (WMMA inner kernels; state slice resident in LDS)
  delta_kernel<<<64, 256, DELTA_SMEM, stream>>>(qc_h, kc_h, vc_f, betab, dob);

  // 6) stream stats + gate input
  stats_kernel<<<Mrows, 256, 0, stream>>>(vc_f, dob, f1, f3, f7, f15, f31, stats);
  gi_kernel<<<blocks((size_t)Mrows * Kgate), 256, 0, stream>>>(hid, stats, gi_h);

  // 7) gate MLP layer 1 (WMMA, fused bias+GELU, f16 out)
  dim3 gg1(Ngate / 64, Mrows / 128);
  gemm_wmma<2><<<gg1, 256, 0, stream>>>(gi_h, gw1_h, gb1, hm_h, Mrows, Ngate, Kgate);

  // 8) gate layer 2 + softplus + normalize
  gate2_kernel<<<Mrows, 256, 0, stream>>>(hm_h, gw2, gb2, wts);

  // 9) combine streams + RMS norm
  combine_kernel<<<Mrows, 256, 0, stream>>>(vc_f, dob, wts, f1, f3, f7, f15,
                                            f31, nw, o_h);

  // 10) output projection (WMMA, f32 out)
  dim3 go(Dd / 64, Mrows / 128);
  gemm_wmma<0><<<go, 256, 0, stream>>>(o_h, wo_h, nullptr, out, Mrows, Dd, Dd);
}
